// MultiHeadAttentionQuantum_65481071402455
// MI455X (gfx1250) — matrix operand order repaired and checked by reading, not compiled, not run on hardware
//
#include <hip/hip_runtime.h>

typedef __attribute__((ext_vector_type(16))) _Float16 v16h;
typedef __attribute__((ext_vector_type(8)))  _Float16 v8h;
typedef __attribute__((ext_vector_type(8)))  float    v8f;

#define K_TOT   1024
#define N_TOT   1024
#define K_STEP  32
#define M_BLK   128
#define N_BLK   256
#define LDT     136   // ActT row stride in halves (272B: 16B aligned, conflict-skewed)

// One-shot W f32 -> f16 conversion into workspace (keeps hot loop free of cvt).
__global__ __launch_bounds__(256) void w_to_f16(const float* __restrict__ W,
                                                _Float16* __restrict__ Wh) {
    const int i = (blockIdx.x * 256 + threadIdx.x) * 8;
    const float4 a = *(const float4*)(W + i);
    const float4 b = *(const float4*)(W + i + 4);
    v8h h;
    h[0] = (_Float16)a.x; h[1] = (_Float16)a.y; h[2] = (_Float16)a.z; h[3] = (_Float16)a.w;
    h[4] = (_Float16)b.x; h[5] = (_Float16)b.y; h[6] = (_Float16)b.z; h[7] = (_Float16)b.w;
    *(v8h*)(Wh + i) = h;
}

// out = cos(x + theta) @ W^T + b, computed as out^T tiles:
//   D(16n x 16m) = Wtile(16n x 32k) * ActT(32k x 16m), f16 in / f32 accum.
// Wave tile: 32m x 128n (2 x 8 WMMA accumulators). Block tile: 128m x 256n.
template <bool PRECONV>
__global__ __launch_bounds__(256) void qmha_cos_gemm_wmma(
    const float*    __restrict__ x,
    const float*    __restrict__ theta,
    const float*    __restrict__ W,
    const _Float16* __restrict__ Wh,
    const float*    __restrict__ bias,
    float*          __restrict__ out)
{
    __shared__ _Float16 ActT[K_STEP * LDT];  // transposed f16 activations: [k][m]
    __shared__ float    thS[K_TOT];          // flattened theta (H*DK = 1024)

    const int tid  = threadIdx.x;
    const int lane = tid & 31;
    const int wave = tid >> 5;
    const int wm   = wave & 3;    // 4 waves along m: 32 rows each
    const int wn   = wave >> 2;   // 2 waves along n: 128 cols each

    const int mBase = blockIdx.x * M_BLK;
    const int nBase = blockIdx.y * N_BLK;

    const int lhalf = lane & 15;  // A-frag: n row within tile; D-frag: m column
    const int lhi   = lane >> 4;  // half-wave select
    const int kb    = lhi << 3;   // A-frag K chunk base per ISA 16-bit A layout

    for (int i = tid; i < K_TOT; i += 256) thS[i] = theta[i];

    v8f acc[2][8] = {};

    for (int k0 = 0; k0 < K_TOT; k0 += K_STEP) {
        __syncthreads();  // also covers thS preload on first iteration

        // Stage ActT[k][m] = (f16) cos(x[m,k0+k] + theta[k0+k]); v_cos_f32 fast path.
        #pragma unroll
        for (int cc = 0; cc < 4; ++cc) {
            const int c  = (tid << 2) + cc;     // 1024 float4-chunks per K-slab
            const int ml = c >> 3;              // 0..127 (m within block)
            const int kc = (c & 7) << 2;        // 0..28  (k within slab)
            const float4 xv = *(const float4*)(x + (size_t)(mBase + ml) * K_TOT + k0 + kc);
            const float r = 0.15915494309189535f;  // 1/(2*pi): v_cos takes revolutions
            ActT[(kc + 0) * LDT + ml] = (_Float16)__builtin_amdgcn_cosf((xv.x + thS[k0 + kc + 0]) * r);
            ActT[(kc + 1) * LDT + ml] = (_Float16)__builtin_amdgcn_cosf((xv.y + thS[k0 + kc + 1]) * r);
            ActT[(kc + 2) * LDT + ml] = (_Float16)__builtin_amdgcn_cosf((xv.z + thS[k0 + kc + 2]) * r);
            ActT[(kc + 3) * LDT + ml] = (_Float16)__builtin_amdgcn_cosf((xv.w + thS[k0 + kc + 3]) * r);
        }
        if (k0 + K_STEP < K_TOT) {  // speculative prefetch of next x K-slab
            __builtin_prefetch(x + (size_t)(mBase + ((tid << 2) >> 3)) * K_TOT + k0 + K_STEP, 0, 1);
        }
        __syncthreads();

        // B-operand fragments from LDS: lane = m column (lhalf), K chunks [kb,kb+8) + [kb+16,kb+24).
        v16h bf[2];
        #pragma unroll
        for (int mt = 0; mt < 2; ++mt)
            #pragma unroll
            for (int s = 0; s < 16; ++s)
                bf[mt][s] = ActT[(kb + (s & 7) + ((s >> 3) << 4)) * LDT + (wm << 5) + (mt << 4) + lhalf];

        #pragma unroll
        for (int nt = 0; nt < 8; ++nt) {
            // A-operand: lane holds W row n = lhalf, K chunks [kb,kb+8) + [kb+16,kb+24).
            const size_t woff = (size_t)(nBase + (wn << 7) + (nt << 4) + lhalf) * K_TOT + k0 + kb;
            v16h af;
            if constexpr (PRECONV) {
                const v8h c0 = *(const v8h*)(Wh + woff);
                const v8h c1 = *(const v8h*)(Wh + woff + 16);
                af = __builtin_shufflevector(c0, c1, 0, 1, 2, 3, 4, 5, 6, 7,
                                             8, 9, 10, 11, 12, 13, 14, 15);
            } else {
                const float4 w0 = *(const float4*)(W + woff);
                const float4 w1 = *(const float4*)(W + woff + 4);
                const float4 w2 = *(const float4*)(W + woff + 16);
                const float4 w3 = *(const float4*)(W + woff + 20);
                af[0]  = (_Float16)w0.x; af[1]  = (_Float16)w0.y; af[2]  = (_Float16)w0.z; af[3]  = (_Float16)w0.w;
                af[4]  = (_Float16)w1.x; af[5]  = (_Float16)w1.y; af[6]  = (_Float16)w1.z; af[7]  = (_Float16)w1.w;
                af[8]  = (_Float16)w2.x; af[9]  = (_Float16)w2.y; af[10] = (_Float16)w2.z; af[11] = (_Float16)w2.w;
                af[12] = (_Float16)w3.x; af[13] = (_Float16)w3.y; af[14] = (_Float16)w3.z; af[15] = (_Float16)w3.w;
            }
            #pragma unroll
            for (int mt = 0; mt < 2; ++mt)
                acc[mt][nt] = __builtin_amdgcn_wmma_f32_16x16x32_f16(
                    false, af, false, bf[mt], (short)0, acc[mt][nt], false, false);
        }
    }

    // Epilogue: D lane layout -> m column = lhalf, VGPR r -> n row = lhi*8 + r,
    // so each lane owns 8 consecutive n in one out row: two b128 stores per tile.
    #pragma unroll
    for (int mt = 0; mt < 2; ++mt) {
        const int row = mBase + (wm << 5) + (mt << 4) + lhalf;
        #pragma unroll
        for (int nt = 0; nt < 8; ++nt) {
            const int ncol = nBase + (wn << 7) + (nt << 4) + (lhi << 3);
            const float4 b0 = *(const float4*)(bias + ncol);
            const float4 b1 = *(const float4*)(bias + ncol + 4);
            float4 o0, o1;
            o0.x = acc[mt][nt][0] + b0.x; o0.y = acc[mt][nt][1] + b0.y;
            o0.z = acc[mt][nt][2] + b0.z; o0.w = acc[mt][nt][3] + b0.w;
            o1.x = acc[mt][nt][4] + b1.x; o1.y = acc[mt][nt][5] + b1.y;
            o1.z = acc[mt][nt][6] + b1.z; o1.w = acc[mt][nt][7] + b1.w;
            *(float4*)(out + (size_t)row * N_TOT + ncol)     = o0;
            *(float4*)(out + (size_t)row * N_TOT + ncol + 4) = o1;
        }
    }
}

extern "C" void kernel_launch(void* const* d_in, const int* in_sizes, int n_in,
                              void* d_out, int out_size, void* d_ws, size_t ws_size,
                              hipStream_t stream) {
    (void)n_in; (void)out_size;
    const float* x     = (const float*)d_in[0];  // [B,S,E] f32
    const float* theta = (const float*)d_in[1];  // [H,DK]  f32 (flat 1024)
    const float* W     = (const float*)d_in[2];  // [E,E]   f32
    const float* b     = (const float*)d_in[3];  // [E]     f32
    float* out = (float*)d_out;                  // [B,S,E] f32

    const int M = in_sizes[0] / K_TOT;           // B*S = 16384
    dim3 grid(M / M_BLK, N_TOT / N_BLK);         // (128, 4)

    const size_t WH_BYTES = (size_t)N_TOT * K_TOT * sizeof(_Float16);  // 2 MB
    if (ws_size >= WH_BYTES) {
        _Float16* Wh = (_Float16*)d_ws;
        w_to_f16<<<(N_TOT * K_TOT) / (256 * 8), 256, 0, stream>>>(W, Wh);
        qmha_cos_gemm_wmma<true><<<grid, 256, 0, stream>>>(x, theta, W, Wh, b, out);
    } else {
        qmha_cos_gemm_wmma<false><<<grid, 256, 0, stream>>>(x, theta, W, nullptr, b, out);
    }
}
